// AttentiveAggregation_35656818491723
// MI455X (gfx1250) — compile-verified, matrix operand order checked by reading, not run on hardware
//
#include <hip/hip_runtime.h>
#include <math.h>

#define D      128
#define NT     8      // 128 / 16 column tiles
#define KC     4      // 128 / 32 k-chunks
#define MAXE   4096   // max rows per segment staged in LDS (mean=61, >500 sigma safe)
#define WPAD   136    // padded W row stride (bf16 elems) -> 272B, bank-conflict free
#define BLOCK  128

typedef __bf16 bf16_t;
typedef bf16_t v16bf __attribute__((ext_vector_type(16)));
typedef float  v8f   __attribute__((ext_vector_type(8)));

union Frag16 {
    v16bf v;
    unsigned short u[16];
    uint4 q[2];
};

__device__ __forceinline__ unsigned short f2bf(float f) {
    // round-to-nearest-even fp32 -> bf16
    unsigned int u = __float_as_uint(f);
    unsigned int r = u + 0x7FFFu + ((u >> 16) & 1u);
    return (unsigned short)(r >> 16);
}

#if defined(__has_builtin)
#if __has_builtin(__builtin_amdgcn_tanhf)
#define TANHF(x) __builtin_amdgcn_tanhf(x)
#else
#define TANHF(x) tanhf(x)
#endif
#else
#define TANHF(x) tanhf(x)
#endif

__global__ __launch_bounds__(BLOCK) void attnagg_fused_kernel(
    const float* __restrict__ H,      // [V,128]
    const int*   __restrict__ batch,  // [V] sorted segment ids
    const float* __restrict__ Wp,     // [128,128]
    const float* __restrict__ bp,     // [128]
    const float* __restrict__ ws,     // [128]
    float*       __restrict__ out,    // [G,128]
    int V)
{
    __shared__ __align__(16) unsigned short Wlds[D * WPAD]; // bf16 W, padded rows
    __shared__ float eLds[MAXE];
    __shared__ float bLds[D];
    __shared__ float wLds[D];
    __shared__ float red[4];
    __shared__ int   bounds[2];

    const int tid  = threadIdx.x;
    const int g    = blockIdx.x;
    const int wave = tid >> 5;
    const int lane = tid & 31;
    const int m    = lane & 15;   // row (A) / col (B) within 16
    const int half = lane >> 4;   // selects K 0-7/16-23 vs 8-15/24-31

    // --- segment bounds via binary search on sorted batch ---
    if (tid == 0) {
        int lo = 0, hi = V;
        while (lo < hi) { int mid = (lo + hi) >> 1; if (batch[mid] < g) lo = mid + 1; else hi = mid; }
        bounds[0] = lo;
        int lo2 = lo, hi2 = V;
        while (lo2 < hi2) { int mid = (lo2 + hi2) >> 1; if (batch[mid] < g + 1) lo2 = mid + 1; else hi2 = mid; }
        bounds[1] = lo2;
    }
    // --- W_proj -> bf16 LDS (row-major, padded stride) ---
    for (int i = tid; i < D * D; i += BLOCK) {
        int n = i >> 7, k = i & (D - 1);
        Wlds[n * WPAD + k] = f2bf(Wp[i]);
    }
    bLds[tid] = bp[tid];  // BLOCK == D
    wLds[tid] = ws[tid];
    __syncthreads();

    const int lo     = bounds[0];
    const int segLen = bounds[1] - lo;
    const int nTiles = (segLen + 15) >> 4;

    // ================= Phase 1: e_v = w . tanh(W h_v + b) via bf16 WMMA =================
    for (int tile = wave; tile < nTiles; tile += 4) {
        const int  rloc  = tile * 16 + m;
        const bool valid = rloc < segLen;
        const float* __restrict__ hrow = H + (size_t)(lo + rloc) * D;

        // A fragments: 16 rows x 128 K of H as bf16 (ISA 16-bit A layout)
        Frag16 A[KC];
#pragma unroll
        for (int kc = 0; kc < KC; ++kc) {
            const int kb = kc * 32 + half * 8;
            if (valid) {
                float4 a0 = *(const float4*)(hrow + kb);
                float4 a1 = *(const float4*)(hrow + kb + 4);
                float4 a2 = *(const float4*)(hrow + kb + 16);
                float4 a3 = *(const float4*)(hrow + kb + 20);
                A[kc].u[0]  = f2bf(a0.x); A[kc].u[1]  = f2bf(a0.y);
                A[kc].u[2]  = f2bf(a0.z); A[kc].u[3]  = f2bf(a0.w);
                A[kc].u[4]  = f2bf(a1.x); A[kc].u[5]  = f2bf(a1.y);
                A[kc].u[6]  = f2bf(a1.z); A[kc].u[7]  = f2bf(a1.w);
                A[kc].u[8]  = f2bf(a2.x); A[kc].u[9]  = f2bf(a2.y);
                A[kc].u[10] = f2bf(a2.z); A[kc].u[11] = f2bf(a2.w);
                A[kc].u[12] = f2bf(a3.x); A[kc].u[13] = f2bf(a3.y);
                A[kc].u[14] = f2bf(a3.z); A[kc].u[15] = f2bf(a3.w);
            } else {
                A[kc].q[0] = make_uint4(0u, 0u, 0u, 0u);
                A[kc].q[1] = make_uint4(0u, 0u, 0u, 0u);
            }
        }

        float s[8] = {0.f, 0.f, 0.f, 0.f, 0.f, 0.f, 0.f, 0.f};
#pragma unroll
        for (int nt = 0; nt < NT; ++nt) {
            const int n = nt * 16 + m;         // this lane's output column
            v8f acc = {0.f, 0.f, 0.f, 0.f, 0.f, 0.f, 0.f, 0.f};
#pragma unroll
            for (int kc = 0; kc < KC; ++kc) {
                const int kb = kc * 32 + half * 8;
                Frag16 B;  // B[k][n] = W[n][k] -> same lane packing as A with n as "row"
                const unsigned short* wrow = &Wlds[n * WPAD + kb];
                B.q[0] = *(const uint4*)(wrow);
                B.q[1] = *(const uint4*)(wrow + 16);
                acc = __builtin_amdgcn_wmma_f32_16x16x32_bf16(
                    false, A[kc].v, false, B.v, (short)0, acc, false, false);
            }
            const float bn = bLds[n];
            const float wn = wLds[n];
#pragma unroll
            for (int gg = 0; gg < 8; ++gg)     // acc[gg] = row (8*half+gg), col n
                s[gg] += wn * TANHF(acc[gg] + bn);
        }
        // reduce over the 16 lanes of each half-group (covers all 128 columns)
#pragma unroll
        for (int off = 1; off < 16; off <<= 1) {
#pragma unroll
            for (int gg = 0; gg < 8; ++gg)
                s[gg] += __shfl_xor(s[gg], off, 32);
        }
        // lane m==gg stores row tile*16 + 8*half + gg
#pragma unroll
        for (int gg = 0; gg < 8; ++gg) {
            if (m == gg) {
                const int ridx = tile * 16 + half * 8 + gg;
                if (ridx < segLen) eLds[ridx] = s[gg];
            }
        }
    }
    __syncthreads();

    // ================= Phase 2: segment softmax in LDS =================
    float mx = -INFINITY;
    for (int v = tid; v < segLen; v += BLOCK) mx = fmaxf(mx, eLds[v]);
#pragma unroll
    for (int off = 16; off > 0; off >>= 1) mx = fmaxf(mx, __shfl_xor(mx, off, 32));
    if (lane == 0) red[wave] = mx;
    __syncthreads();
    mx = fmaxf(fmaxf(red[0], red[1]), fmaxf(red[2], red[3]));
    __syncthreads();

    float sm = 0.f;
    for (int v = tid; v < segLen; v += BLOCK) {
        float a = __expf(eLds[v] - mx);
        eLds[v] = a;
        sm += a;
    }
#pragma unroll
    for (int off = 16; off > 0; off >>= 1) sm += __shfl_xor(sm, off, 32);
    if (lane == 0) red[wave] = sm;
    __syncthreads();
    sm = red[0] + red[1] + red[2] + red[3];
    const float inv = 1.0f / fmaxf(sm, 1e-12f);

    // ================= Phase 3: pooled output, thread = column d =================
    // segment's H rows were just streamed -> L2/WGP$-hot second pass
    float accD = 0.f;
    const float* __restrict__ hbase = H + (size_t)lo * D + tid;
    for (int v = 0; v < segLen; ++v)
        accD = fmaf(eLds[v], hbase[(size_t)v * D], accD);
    out[(size_t)g * D + tid] = accD * inv;
}

extern "C" void kernel_launch(void* const* d_in, const int* in_sizes, int n_in,
                              void* d_out, int out_size, void* d_ws, size_t ws_size,
                              hipStream_t stream) {
    const float* H     = (const float*)d_in[0];
    const int*   batch = (const int*)d_in[1];
    const float* Wp    = (const float*)d_in[2];
    const float* bp    = (const float*)d_in[3];
    const float* ws    = (const float*)d_in[4];
    float* out = (float*)d_out;

    const int V = in_sizes[1];
    const int G = out_size / D;   // 16384 segments

    hipLaunchKernelGGL(attnagg_fused_kernel, dim3(G), dim3(BLOCK), 0, stream,
                       H, batch, Wp, bp, ws, out, V);
}
